// SparseMLP_36996848288082
// MI455X (gfx1250) — compile-verified, matrix operand order checked by reading, not run on hardware
//
#include <hip/hip_runtime.h>
#include <cstdint>

// Problem constants (from reference setup_inputs)
#define TOK   8192          // B*S tokens
#define HDIM  2048
#define IDIM  8192
#define NEXP  8
#define CAP   1280          // floor(1.25*8192/8), even

// GEMM tiling
#define BM 128
#define BN 128
#define BK 32
#define LDSS 40             // padded LDS row stride in bf16 elements (80B, 16B aligned)

typedef __attribute__((ext_vector_type(16))) __bf16 v16bf;
typedef __attribute__((ext_vector_type(8)))  __bf16 v8bf;
typedef __attribute__((ext_vector_type(4)))  __bf16 v4bf;
typedef __attribute__((ext_vector_type(8)))  float  v8f;
typedef __attribute__((ext_vector_type(4)))  float  f32x4;

union V16 { v16bf v; v8bf h[2]; };

// ---- CDNA5 async load-to-LDS (ASYNCcnt) support, with safe fallback -------
#if __has_builtin(__builtin_amdgcn_global_load_async_to_lds_b128)
#define HAVE_ASYNC_LDS 1
// builtin signature (from hipcc diagnostic): (v4i AS1*, v4i AS3*, Imm int, Imm int)
typedef int v4i_vs __attribute__((vector_size(16)));
typedef v4i_vs __attribute__((address_space(1))) gv4i;
typedef v4i_vs __attribute__((address_space(3))) lv4i;
__device__ __forceinline__ void async_copy16(const __bf16* g, __bf16* l) {
    __builtin_amdgcn_global_load_async_to_lds_b128((gv4i*)g, (lv4i*)l, 0, 0);
}
#else
#define HAVE_ASYNC_LDS 0
#endif

#if __has_builtin(__builtin_amdgcn_s_wait_asynccnt)
#define WAIT_ASYNC0() __builtin_amdgcn_s_wait_asynccnt(0)
#else
#define WAIT_ASYNC0() asm volatile("s_wait_asynccnt 0x0" ::: "memory")
#endif

// ---------------------------------------------------------------------------
// Kernel 1: gating — one wave (32 lanes) per token.
// ---------------------------------------------------------------------------
__global__ void __launch_bounds__(256) gate_kernel(
    const float* __restrict__ tokens,     // (TOK, HDIM)
    const float* __restrict__ gw,         // (NEXP, HDIM)
    int* __restrict__ top1,               // (TOK)
    float* __restrict__ prob)             // (TOK) prob of top1 expert
{
    const int t    = (blockIdx.x * blockDim.x + threadIdx.x) >> 5;
    const int lane = threadIdx.x & 31;
    if (t >= TOK) return;
    const float* trow = tokens + (size_t)t * HDIM;

    float acc[NEXP];
#pragma unroll
    for (int e = 0; e < NEXP; ++e) acc[e] = 0.f;

    for (int h = lane; h < HDIM; h += 32) {
        float x = trow[h];
#pragma unroll
        for (int e = 0; e < NEXP; ++e) acc[e] += x * gw[e * HDIM + h];
    }
#pragma unroll
    for (int off = 16; off > 0; off >>= 1) {
#pragma unroll
        for (int e = 0; e < NEXP; ++e) acc[e] += __shfl_xor(acc[e], off, 32);
    }
    float m = acc[0]; int bi = 0;
#pragma unroll
    for (int e = 1; e < NEXP; ++e) { if (acc[e] > m) { m = acc[e]; bi = e; } }  // first-max
    float s = 0.f;
#pragma unroll
    for (int e = 0; e < NEXP; ++e) s += __expf(acc[e] - m);
    if (lane == 0) { top1[t] = bi; prob[t] = 1.0f / s; }
}

// ---------------------------------------------------------------------------
// Kernel 2: ordered capacity routing (single workgroup prefix scan).
// ---------------------------------------------------------------------------
__global__ void __launch_bounds__(1024) scan_kernel(
    const int*   __restrict__ top1,
    const float* __restrict__ prob,
    int*   __restrict__ slot_tok,         // (NEXP*CAP), -1 = empty
    float* __restrict__ slot_w)           // (NEXP*CAP)
{
    __shared__ int warp_tot[32][NEXP];
    const int tid  = threadIdx.x;
    const int lane = tid & 31;
    const int w    = tid >> 5;

    for (int i = tid; i < NEXP * CAP; i += 1024) { slot_tok[i] = -1; slot_w[i] = 0.f; }
    __syncthreads();

    const int base = tid * 8;
    int idx[8];
    int cnt[NEXP];
#pragma unroll
    for (int e = 0; e < NEXP; ++e) cnt[e] = 0;
#pragma unroll
    for (int j = 0; j < 8; ++j) {
        idx[j] = top1[base + j];
#pragma unroll
        for (int e = 0; e < NEXP; ++e) cnt[e] += (idx[j] == e) ? 1 : 0;
    }
    int scan[NEXP];
#pragma unroll
    for (int e = 0; e < NEXP; ++e) scan[e] = cnt[e];
#pragma unroll
    for (int off = 1; off < 32; off <<= 1) {
        int v[NEXP];
#pragma unroll
        for (int e = 0; e < NEXP; ++e) v[e] = __shfl_up(scan[e], off, 32);
        if (lane >= off) {
#pragma unroll
            for (int e = 0; e < NEXP; ++e) scan[e] += v[e];
        }
    }
    if (lane == 31) {
#pragma unroll
        for (int e = 0; e < NEXP; ++e) warp_tot[w][e] = scan[e];
    }
    __syncthreads();
    if (tid < 32) {
        int v[NEXP];
#pragma unroll
        for (int e = 0; e < NEXP; ++e) v[e] = warp_tot[tid][e];
#pragma unroll
        for (int off = 1; off < 32; off <<= 1) {
            int u[NEXP];
#pragma unroll
            for (int e = 0; e < NEXP; ++e) u[e] = __shfl_up(v[e], off, 32);
            if (tid >= off) {
#pragma unroll
                for (int e = 0; e < NEXP; ++e) v[e] += u[e];
            }
        }
#pragma unroll
        for (int e = 0; e < NEXP; ++e) warp_tot[tid][e] = v[e];
    }
    __syncthreads();

    int run[NEXP];
#pragma unroll
    for (int e = 0; e < NEXP; ++e)
        run[e] = scan[e] - cnt[e] + ((w > 0) ? warp_tot[w - 1][e] : 0);

#pragma unroll
    for (int j = 0; j < 8; ++j) {
        const int t = base + j;
        int r = 0, ex = idx[j];
#pragma unroll
        for (int e = 0; e < NEXP; ++e) {
            if (ex == e) { r = run[e]; run[e] = r + 1; }
        }
        if (r < CAP) {
            const int slot = ex * CAP + r;
            slot_tok[slot] = t;
            slot_w[slot]   = prob[t];
        }
    }
}

// ---------------------------------------------------------------------------
// Kernel 3: GEMM1 fused gather + exact GELU. Per expert: (CAP x HDIM) x
// (HDIM x IDIM) -> h bf16. A rows gathered from tokens via slot map.
// ---------------------------------------------------------------------------
__global__ void __launch_bounds__(256) moe_gemm1(
    const float* __restrict__ tokens,     // (TOK, HDIM)
    const float* __restrict__ wi,         // (NEXP, HDIM, IDIM)
    const int*   __restrict__ slot_tok,   // (NEXP*CAP)
    __bf16* __restrict__ hbuf)            // (NEXP, CAP, IDIM) bf16
{
    const int e   = blockIdx.z;
    const int nb  = blockIdx.x * BN;
    const int mb  = blockIdx.y * BM;
    const int tid = threadIdx.x;
    const int lane = tid & 31, wv = tid >> 5;
    const int wm = wv & 1, wn = wv >> 1;   // 2 x 4 wave grid; wave tile 64(M) x 32(N)

    __shared__ __bf16 As[BM * LDSS];
    __shared__ __bf16 Bs[BN * LDSS];

    // A loader: 2 threads per row, 16 floats each
    const int arow = tid >> 1;
    const int acol = (tid & 1) * 16;
    const int tok  = slot_tok[e * CAP + mb + arow];
    const float* aptr = tokens + (size_t)((tok >= 0) ? tok : 0) * HDIM + acol;
    const bool avalid = (tok >= 0);

    // B loader: 4 K-rows x 4 N-cols per thread; global reads coalesced along N,
    // LDS stores packed to ds_store_b64 (4 K-adjacent bf16 in the Bs[n][k] layout)
    const int k0 = (tid >> 5) * 4;        // 8 k-groups
    const int n0 = (tid & 31) * 4;        // 32 n-groups
    const float* bptr = wi + (size_t)e * HDIM * IDIM + (size_t)k0 * IDIM + nb + n0;

    v8f acc[4][2];
    const v8f z8 = {0.f, 0.f, 0.f, 0.f, 0.f, 0.f, 0.f, 0.f};
#pragma unroll
    for (int i = 0; i < 4; ++i)
#pragma unroll
        for (int j = 0; j < 2; ++j) acc[i][j] = z8;

    f32x4 areg[4], breg[4];
    const f32x4 z4 = {0.f, 0.f, 0.f, 0.f};
#pragma unroll
    for (int j = 0; j < 4; ++j) {
        areg[j] = avalid ? *(const f32x4*)(aptr + j * 4) : z4;
        breg[j] = *(const f32x4*)(bptr + (size_t)j * IDIM);
    }

    const int NK = HDIM / BK;
    for (int kt = 0; kt < NK; ++kt) {
        if (kt) __syncthreads();          // previous compute done reading LDS
        // A regs -> LDS (fp32->bf16), merged b128 stores
        {
            v8bf a0, a1;
#pragma unroll
            for (int q = 0; q < 8; ++q) a0[q] = (__bf16)areg[q >> 2][q & 3];
#pragma unroll
            for (int q = 0; q < 8; ++q) a1[q] = (__bf16)areg[2 + (q >> 2)][q & 3];
            *(v8bf*)(As + arow * LDSS + acol)     = a0;
            *(v8bf*)(As + arow * LDSS + acol + 8) = a1;
        }
        // B regs -> LDS transposed: Bs[n][k], packed b64 stores
#pragma unroll
        for (int j = 0; j < 4; ++j) {
            v4bf p;
            p[0] = (__bf16)breg[0][j];
            p[1] = (__bf16)breg[1][j];
            p[2] = (__bf16)breg[2][j];
            p[3] = (__bf16)breg[3][j];
            *(v4bf*)(Bs + (n0 + j) * LDSS + k0) = p;
        }
        if (kt + 1 < NK) {
            const size_t ka = (size_t)(kt + 1) * BK;
#pragma unroll
            for (int j = 0; j < 4; ++j) {
                areg[j] = avalid ? *(const f32x4*)(aptr + ka + j * 4) : z4;
                breg[j] = *(const f32x4*)(bptr + (ka + j) * IDIM);
            }
        }
        __syncthreads();

        const int lr = lane & 15, kh = lane >> 4;
        V16 afrag[4], bfrag[2];
#pragma unroll
        for (int mi = 0; mi < 4; ++mi) {
            const int b = (wm * 64 + mi * 16 + lr) * LDSS + kh * 8;
            afrag[mi].h[0] = *(const v8bf*)(As + b);
            afrag[mi].h[1] = *(const v8bf*)(As + b + 16);
        }
#pragma unroll
        for (int ni = 0; ni < 2; ++ni) {
            const int b = (wn * 32 + ni * 16 + lr) * LDSS + kh * 8;
            bfrag[ni].h[0] = *(const v8bf*)(Bs + b);
            bfrag[ni].h[1] = *(const v8bf*)(Bs + b + 16);
        }
#pragma unroll
        for (int mi = 0; mi < 4; ++mi)
#pragma unroll
            for (int ni = 0; ni < 2; ++ni)
                acc[mi][ni] = __builtin_amdgcn_wmma_f32_16x16x32_bf16(
                    false, afrag[mi].v, false, bfrag[ni].v,
                    (short)0, acc[mi][ni], false, false);
    }

    // epilogue: exact GELU, store bf16
    const int lr = lane & 15, kh = lane >> 4;
#pragma unroll
    for (int mi = 0; mi < 4; ++mi) {
#pragma unroll
        for (int ni = 0; ni < 2; ++ni) {
            const int n = nb + wn * 32 + ni * 16 + lr;
#pragma unroll
            for (int v = 0; v < 8; ++v) {
                const int m = mb + wm * 64 + mi * 16 + kh * 8 + v;
                const float x = acc[mi][ni][v];
                const float g = 0.5f * x * (1.0f + erff(x * 0.70710678118654752440f));
                hbuf[((size_t)e * CAP + m) * IDIM + n] = (__bf16)g;
            }
        }
    }
}

// ---------------------------------------------------------------------------
// Kernel 4: GEMM2 fused combine/scatter. A tile (bf16->bf16) streamed with
// CDNA5 async load-to-LDS (double-buffered, ASYNCcnt synchronized).
// ---------------------------------------------------------------------------
__global__ void __launch_bounds__(256) moe_gemm2(
    const __bf16* __restrict__ hbuf,      // (NEXP, CAP, IDIM) bf16
    const float*  __restrict__ wo,        // (NEXP, IDIM, HDIM)
    const int*    __restrict__ slot_tok,
    const float*  __restrict__ slot_w,
    float* __restrict__ out)              // (TOK, HDIM), pre-zeroed
{
    const int e   = blockIdx.z;
    const int nb  = blockIdx.x * BN;
    const int mb  = blockIdx.y * BM;
    const int tid = threadIdx.x;
    const int lane = tid & 31, wv = tid >> 5;
    const int wm = wv & 1, wn = wv >> 1;

    __shared__ __bf16 As[2][BM * LDSS];
    __shared__ __bf16 Bs[BN * LDSS];

    const int arow = tid >> 1;
    const int acol = (tid & 1) * 16;
    const __bf16* aptr = hbuf + ((size_t)e * CAP + mb + arow) * IDIM + acol;

    const int k0 = (tid >> 5) * 4;
    const int n0 = (tid & 31) * 4;
    const float* bptr = wo + (size_t)e * IDIM * HDIM + (size_t)k0 * HDIM + nb + n0;

    v8f acc[4][2];
    const v8f z8 = {0.f, 0.f, 0.f, 0.f, 0.f, 0.f, 0.f, 0.f};
#pragma unroll
    for (int i = 0; i < 4; ++i)
#pragma unroll
        for (int j = 0; j < 2; ++j) acc[i][j] = z8;

    f32x4 breg[4];
#pragma unroll
    for (int j = 0; j < 4; ++j) breg[j] = *(const f32x4*)(bptr + (size_t)j * HDIM);

#if HAVE_ASYNC_LDS
    // kick off tile 0 into buffer 0
    async_copy16(aptr,     &As[0][arow * LDSS + acol]);
    async_copy16(aptr + 8, &As[0][arow * LDSS + acol + 8]);
#else
    v8bf areg0 = *(const v8bf*)(aptr);
    v8bf areg1 = *(const v8bf*)(aptr + 8);
#endif

    const int NK = IDIM / BK;
    for (int kt = 0; kt < NK; ++kt) {
        const int cur = kt & 1;
        if (kt) __syncthreads();          // previous compute done reading LDS
#if HAVE_ASYNC_LDS
        WAIT_ASYNC0();                    // As[cur] writes (issued last iter) complete
        if (kt + 1 < NK) {
            const __bf16* src = aptr + (size_t)(kt + 1) * BK;
            async_copy16(src,     &As[cur ^ 1][arow * LDSS + acol]);
            async_copy16(src + 8, &As[cur ^ 1][arow * LDSS + acol + 8]);
        }
#else
        *(v8bf*)(&As[cur][arow * LDSS + acol])     = areg0;
        *(v8bf*)(&As[cur][arow * LDSS + acol + 8]) = areg1;
        if (kt + 1 < NK) {
            const size_t ka = (size_t)(kt + 1) * BK;
            areg0 = *(const v8bf*)(aptr + ka);
            areg1 = *(const v8bf*)(aptr + ka + 8);
        }
#endif
        // B regs -> LDS transposed, packed b64 stores
#pragma unroll
        for (int j = 0; j < 4; ++j) {
            v4bf p;
            p[0] = (__bf16)breg[0][j];
            p[1] = (__bf16)breg[1][j];
            p[2] = (__bf16)breg[2][j];
            p[3] = (__bf16)breg[3][j];
            *(v4bf*)(Bs + (n0 + j) * LDSS + k0) = p;
        }
        if (kt + 1 < NK) {
            const size_t ka = (size_t)(kt + 1) * BK;
#pragma unroll
            for (int j = 0; j < 4; ++j)
                breg[j] = *(const f32x4*)(bptr + (ka + j) * HDIM);
        }
        __syncthreads();

        const int lr = lane & 15, kh = lane >> 4;
        V16 afrag[4], bfrag[2];
#pragma unroll
        for (int mi = 0; mi < 4; ++mi) {
            const int b = (wm * 64 + mi * 16 + lr) * LDSS + kh * 8;
            afrag[mi].h[0] = *(const v8bf*)(&As[cur][b]);
            afrag[mi].h[1] = *(const v8bf*)(&As[cur][b + 16]);
        }
#pragma unroll
        for (int ni = 0; ni < 2; ++ni) {
            const int b = (wn * 32 + ni * 16 + lr) * LDSS + kh * 8;
            bfrag[ni].h[0] = *(const v8bf*)(Bs + b);
            bfrag[ni].h[1] = *(const v8bf*)(Bs + b + 16);
        }
#pragma unroll
        for (int mi = 0; mi < 4; ++mi)
#pragma unroll
            for (int ni = 0; ni < 2; ++ni)
                acc[mi][ni] = __builtin_amdgcn_wmma_f32_16x16x32_bf16(
                    false, afrag[mi].v, false, bfrag[ni].v,
                    (short)0, acc[mi][ni], false, false);
    }

    // epilogue: scatter rows to token rows scaled by combine weight
    const int lr = lane & 15, kh = lane >> 4;
#pragma unroll
    for (int mi = 0; mi < 4; ++mi) {
        const int m0 = mb + wm * 64 + mi * 16 + kh * 8;
        int   tk[8];
        float wt[8];
#pragma unroll
        for (int v = 0; v < 8; ++v) {
            tk[v] = slot_tok[e * CAP + m0 + v];
            wt[v] = slot_w[e * CAP + m0 + v];
        }
#pragma unroll
        for (int ni = 0; ni < 2; ++ni) {
            const int n = nb + wn * 32 + ni * 16 + lr;
#pragma unroll
            for (int v = 0; v < 8; ++v) {
                if (tk[v] >= 0)
                    out[(size_t)tk[v] * HDIM + n] = acc[mi][ni][v] * wt[v];
            }
        }
    }
}

// ---------------------------------------------------------------------------
extern "C" void kernel_launch(void* const* d_in, const int* in_sizes, int n_in,
                              void* d_out, int out_size, void* d_ws, size_t ws_size,
                              hipStream_t stream) {
    const float* tokens = (const float*)d_in[0];   // inputs (B,S,H) fp32
    const float* gw     = (const float*)d_in[1];   // gate_weight (E,H)
    const float* wi     = (const float*)d_in[2];   // (E,H,I)
    const float* wo     = (const float*)d_in[3];   // (E,I,H)
    float* out = (float*)d_out;

    char* ws = (char*)d_ws;
    int*   top1     = (int*)ws;    ws += (size_t)TOK * sizeof(int);
    float* prob     = (float*)ws;  ws += (size_t)TOK * sizeof(float);
    int*   slot_tok = (int*)ws;    ws += (size_t)NEXP * CAP * sizeof(int);
    float* slot_w   = (float*)ws;  ws += (size_t)NEXP * CAP * sizeof(float);
    ws = (char*)(((uintptr_t)ws + 255) & ~(uintptr_t)255);
    __bf16* hbuf    = (__bf16*)ws; // (NEXP, CAP, IDIM) bf16 = ~168 MB

    (void)hipMemsetAsync(d_out, 0, (size_t)out_size * sizeof(float), stream);

    gate_kernel<<<TOK / 8, 256, 0, stream>>>(tokens, gw, top1, prob);
    scan_kernel<<<1, 1024, 0, stream>>>(top1, prob, slot_tok, slot_w);
    moe_gemm1<<<dim3(IDIM / BN, CAP / BM, NEXP), 256, 0, stream>>>(tokens, wi, slot_tok, hbuf);
    moe_gemm2<<<dim3(HDIM / BN, CAP / BM, NEXP), 256, 0, stream>>>(hbuf, wo, slot_tok, slot_w, out);
}
